// PTBModel_10084583211160
// MI455X (gfx1250) — matrix operand order repaired and checked by reading, not compiled, not run on hardware
//
#include <hip/hip_runtime.h>
#include <hip/hip_bf16.h>

typedef __bf16 bf16;
typedef __attribute__((ext_vector_type(16))) __bf16 v16bf;
typedef __attribute__((ext_vector_type(8)))  float  v8f;

#define VOCAB 10000
#define EMB   1500
#define HID   1500
#define TT    35
#define BB    64
#define ROWS  (TT*BB)        // 2240
#define KP    1504           // EMB/HID padded to mult of 32
#define KP2   3008           // 2*HID padded
#define N0    6144           // 4*HID (6000) padded to mult of 256
#define NV    10240          // VOCAB padded to mult of 256
#define NSLICE 4             // split-K factor for the recurrent GEMMs

#define ALD   40             // padded A-tile row (bf16 elems): 80B, 16B aligned
#define BLD   268            // padded packed-B row (u32): bank-offsets rows by 12

static __device__ __forceinline__ float sigmoidf_(float x) {
    return 1.0f / (1.0f + __expf(-x));
}

// ---------------------------------------------------------------------------
// f32 -> bf16 row-wise convert with destination padding (pad stays 0 via memset)
// ---------------------------------------------------------------------------
__global__ __launch_bounds__(256)
void convert_pad(const float* __restrict__ src, int srcld, int srow0,
                 int ncols, bf16* __restrict__ dst, int dstld, int drow0)
{
    int r = blockIdx.x;
    const float* s = src + (size_t)(srow0 + r) * srcld;
    bf16*        d = dst + (size_t)(drow0 + r) * dstld;
    for (int c = threadIdx.x; c < ncols; c += 256)
        d[c] = (bf16)s[c];
}

// ---------------------------------------------------------------------------
// embedding gather -> bf16 padded rows
// ---------------------------------------------------------------------------
__global__ __launch_bounds__(256)
void embed_kernel(const int* __restrict__ x, const float* __restrict__ emb,
                  bf16* __restrict__ Xb)
{
    int row = blockIdx.x;                 // 0..2239  (t*B + b)
    int xi  = x[row];
    const float* s = emb + (size_t)xi * EMB;
    bf16*        d = Xb  + (size_t)row * KP;
    for (int e = threadIdx.x; e < EMB; e += 256)
        d[e] = (bf16)s[e];
}

// ---------------------------------------------------------------------------
// bf16 GEMM, f32 accumulate:  C(slice)[M,N] = A[M, kslice] @ B[kslice, N]
//  - block = 256 threads = 8 waves; block tile 64x256; wave tile 32x64 (2x4 WMMA)
//  - K stepped by 32 through LDS, double-buffered (1 barrier / step)
//  - B staged PAIR-PACKED via v_perm_b32: BsP[k/2][n] u32 = {K=2j | K=2j+1}
//    so a B fragment is 8x ds_load_b32 (no sub-word gathers)
//  - blockIdx.z = K slice (split-K); partials summed in the consumer kernel
// ---------------------------------------------------------------------------
union AFrag { v16bf v; uint4 q[2]; };
union BFrag { v16bf v; unsigned u[8]; };

__global__ __launch_bounds__(256)
void gemm_bf16_wmma(const bf16* __restrict__ A, int lda,
                    const bf16* __restrict__ Bm, int ldb,
                    float* __restrict__ C, int ldc,
                    int K, int kTilesPerSlice, size_t cSliceStride)
{
    __shared__ __align__(16) bf16     As[2][64][ALD];   // 2 x 5.0 KB
    __shared__ __align__(16) unsigned BsP[2][16][BLD];  // 2 x 16.75 KB

    const int nb = blockIdx.x * 256;
    const int mb = blockIdx.y * 64;
    const int kb = blockIdx.z * kTilesPerSlice * 32;
    int ke = kb + kTilesPerSlice * 32;
    if (ke > K) ke = K;
    C += (size_t)blockIdx.z * cSliceStride;

    const unsigned tid  = threadIdx.x;
    const unsigned lane = tid & 31u;
    const unsigned half = lane >> 4;       // K-half select per ISA lane layout
    const unsigned lm   = lane & 15u;
    const unsigned wave = tid >> 5;
    const unsigned wm   = wave >> 2;       // 0..1  (M sub-tile)
    const unsigned wn   = wave & 3u;       // 0..3  (N sub-tile)

    // cooperative-load coordinates
    const unsigned ar_r  = tid >> 2, ar_q = tid & 3u;   // A: 1 uint4 / thread
    const unsigned bu_k2 = tid >> 5;                    // B unit0 pair-row 0..7
    const unsigned bu_oct= tid & 31u;                   // B col octet

    uint4 aReg;
    uint4 bReg[2][2];   // [unit][k parity]

    auto globalLoad = [&](int k0) {
        aReg = *(const uint4*)(A + (size_t)(mb + ar_r) * lda + k0 + ar_q * 8);
        #pragma unroll
        for (int u = 0; u < 2; ++u) {
            const int k2 = (int)bu_k2 + u * 8;
            const bf16* bp = Bm + (size_t)(k0 + 2 * k2) * ldb + nb + bu_oct * 8;
            bReg[u][0] = *(const uint4*)bp;
            bReg[u][1] = *(const uint4*)(bp + ldb);
        }
    };
    auto ldsStore = [&](int buf) {
        *(uint4*)&As[buf][ar_r][ar_q * 8] = aReg;
        #pragma unroll
        for (int u = 0; u < 2; ++u) {
            // v_perm_b32: S0 supplies bytes 4-7, S1 bytes 0-3 of {S0,S1}
            //   perm(odd, even, 0x05040100) = even.lo16 | odd.lo16 << 16
            //   perm(odd, even, 0x07060302) = even.hi16 | odd.hi16 << 16
            const unsigned* e = (const unsigned*)&bReg[u][0];
            const unsigned* o = (const unsigned*)&bReg[u][1];
            uint4 q0, q1;
            q0.x = __builtin_amdgcn_perm(o[0], e[0], 0x05040100u);
            q0.y = __builtin_amdgcn_perm(o[0], e[0], 0x07060302u);
            q0.z = __builtin_amdgcn_perm(o[1], e[1], 0x05040100u);
            q0.w = __builtin_amdgcn_perm(o[1], e[1], 0x07060302u);
            q1.x = __builtin_amdgcn_perm(o[2], e[2], 0x05040100u);
            q1.y = __builtin_amdgcn_perm(o[2], e[2], 0x07060302u);
            q1.z = __builtin_amdgcn_perm(o[3], e[3], 0x05040100u);
            q1.w = __builtin_amdgcn_perm(o[3], e[3], 0x07060302u);
            unsigned* d = &BsP[buf][bu_k2 + u * 8][bu_oct * 8];
            ((uint4*)d)[0] = q0;
            ((uint4*)d)[1] = q1;
        }
    };

    v8f acc[2][4];
    #pragma unroll
    for (int mi = 0; mi < 2; ++mi)
        #pragma unroll
        for (int ni = 0; ni < 4; ++ni)
            #pragma unroll
            for (int r = 0; r < 8; ++r)
                acc[mi][ni][r] = 0.0f;

    globalLoad(kb);
    ldsStore(0);
    __syncthreads();

    const int ntiles = (ke - kb) >> 5;
    for (int it = 0; it < ntiles; ++it) {
        const int buf = it & 1;
        if (it + 1 < ntiles) globalLoad(kb + (it + 1) * 32);

        // A frags: lanes 0-15: M=lane, elems 0..15 = K0..15;
        //          lanes16-31: M=lane-16, K16..31
        AFrag a[2];
        #pragma unroll
        for (int mi = 0; mi < 2; ++mi) {
            const bf16* ap = &As[buf][wm * 32 + mi * 16 + lm][0];
            a[mi].q[0] = *(const uint4*)(ap + half * 16);
            a[mi].q[1] = *(const uint4*)(ap + half * 16 + 8);
        }
        // B frags: VGPR j = packed pair (K = half*16 + 2j, 2j+1) for N = col
        #pragma unroll
        for (int ni = 0; ni < 4; ++ni) {
            BFrag bfr;
            const unsigned col = wn * 64 + ni * 16 + lm;
            #pragma unroll
            for (int j = 0; j < 8; ++j)
                bfr.u[j] = BsP[buf][half * 8 + j][col];
            #pragma unroll
            for (int mi = 0; mi < 2; ++mi)
                acc[mi][ni] = __builtin_amdgcn_wmma_f32_16x16x32_bf16(
                    false, a[mi].v, false, bfr.v,
                    (short)0, acc[mi][ni], false, false);
        }

        if (it + 1 < ntiles) ldsStore(buf ^ 1);
        __syncthreads();
    }

    // C/D layout: VGPR r -> row (half*8 + r), col = lm
    #pragma unroll
    for (int mi = 0; mi < 2; ++mi)
        #pragma unroll
        for (int ni = 0; ni < 4; ++ni)
            #pragma unroll
            for (int r = 0; r < 8; ++r)
                C[(size_t)(mb + wm * 32 + mi * 16 + half * 8 + r) * ldc
                  + nb + wn * 64 + ni * 16 + lm] = acc[mi][ni][r];
}

// ---------------------------------------------------------------------------
// LSTM cell elementwise: z = sum of split-K partials [+ precomputed x-part] + bias
// gate order i, j, f, o ; forget bias = 1.0
// ---------------------------------------------------------------------------
__global__ __launch_bounds__(256)
void lstm_cell(const float* __restrict__ z, int ldz, int nslices, size_t zSlice,
               const float* __restrict__ pre, int ldpre,
               const float* __restrict__ bias,
               float* __restrict__ c,
               bf16* __restrict__ h_out, int ldh,
               bf16* __restrict__ h_copy, int ldcopy)
{
    int idx = blockIdx.x * 256 + threadIdx.x;
    if (idx >= BB * HID) return;
    int b = idx / HID, u = idx - b * HID;

    float g[4];
    #pragma unroll
    for (int gi = 0; gi < 4; ++gi) {
        int col = gi * HID + u;
        float v = bias[col];
        for (int s = 0; s < nslices; ++s)
            v += z[(size_t)s * zSlice + (size_t)b * ldz + col];
        if (pre) v += pre[(size_t)b * ldpre + col];
        g[gi] = v;
    }
    float cn = c[idx] * sigmoidf_(g[2] + 1.0f) + sigmoidf_(g[0]) * tanhf(g[1]);
    float h  = tanhf(cn) * sigmoidf_(g[3]);
    c[idx] = cn;
    h_out[(size_t)b * ldh + u] = (bf16)h;
    if (h_copy) h_copy[(size_t)b * ldcopy + u] = (bf16)h;
}

// ---------------------------------------------------------------------------
// per-row NLL: rowloss = logsumexp(logits+b) - (logits[tgt]+b[tgt])
// ---------------------------------------------------------------------------
__global__ __launch_bounds__(256)
void loss_row_kernel(const float* __restrict__ logits, int ldl,
                     const float* __restrict__ sb, const int* __restrict__ y,
                     float* __restrict__ rowloss)
{
    __shared__ float red[256];
    int row = blockIdx.x;
    const float* lr = logits + (size_t)row * ldl;

    float mx = -3.0e38f;
    for (int n = threadIdx.x; n < VOCAB; n += 256)
        mx = fmaxf(mx, lr[n] + sb[n]);
    red[threadIdx.x] = mx; __syncthreads();
    for (int s = 128; s > 0; s >>= 1) {
        if (threadIdx.x < (unsigned)s)
            red[threadIdx.x] = fmaxf(red[threadIdx.x], red[threadIdx.x + s]);
        __syncthreads();
    }
    mx = red[0]; __syncthreads();

    float sum = 0.0f;
    for (int n = threadIdx.x; n < VOCAB; n += 256)
        sum += __expf(lr[n] + sb[n] - mx);
    red[threadIdx.x] = sum; __syncthreads();
    for (int s = 128; s > 0; s >>= 1) {
        if (threadIdx.x < (unsigned)s)
            red[threadIdx.x] += red[threadIdx.x + s];
        __syncthreads();
    }
    if (threadIdx.x == 0) {
        int tgt = y[row];
        rowloss[row] = (mx + __logf(red[0])) - (lr[tgt] + sb[tgt]);
    }
}

__global__ __launch_bounds__(256)
void final_reduce(const float* __restrict__ rowloss, float* __restrict__ out)
{
    __shared__ float red[256];
    float s = 0.0f;
    for (int i = threadIdx.x; i < ROWS; i += 256) s += rowloss[i];
    red[threadIdx.x] = s; __syncthreads();
    for (int st = 128; st > 0; st >>= 1) {
        if (threadIdx.x < (unsigned)st) red[threadIdx.x] += red[threadIdx.x + st];
        __syncthreads();
    }
    if (threadIdx.x == 0) out[0] = red[0] / (float)BB;
}

// ---------------------------------------------------------------------------
extern "C" void kernel_launch(void* const* d_in, const int* in_sizes, int n_in,
                              void* d_out, int out_size, void* d_ws, size_t ws_size,
                              hipStream_t stream)
{
    const int*   x   = (const int*)d_in[0];
    const int*   y   = (const int*)d_in[1];
    const float* emb = (const float*)d_in[2];
    const float* W0  = (const float*)d_in[3];
    const float* b0  = (const float*)d_in[4];
    const float* W1  = (const float*)d_in[5];
    const float* b1  = (const float*)d_in[6];
    const float* Wsm = (const float*)d_in[7];
    const float* bsm = (const float*)d_in[8];
    float* out = (float*)d_out;

    char* ws = (char*)d_ws;
    size_t off = 0;
    auto alloc = [&](size_t bytes) -> char* {
        char* p = ws + off;
        off = (off + bytes + 255) & ~(size_t)255;
        return p;
    };
    bf16*  Xb     = (bf16*) alloc((size_t)ROWS * KP  * 2);          // embedded inputs
    bf16*  W0xb   = (bf16*) alloc((size_t)KP   * N0  * 2);          // W0 x-part
    bf16*  W0hb   = (bf16*) alloc((size_t)KP   * N0  * 2);          // W0 h-part
    bf16*  W1cb   = (bf16*) alloc((size_t)KP2  * N0  * 2);          // W1 [h0;h1]
    bf16*  Wsmb   = (bf16*) alloc((size_t)KP   * NV  * 2);          // softmax weights
    float* xW0    = (float*)alloc((size_t)ROWS * N0  * 4);          // precomputed x@W0x
    float* z0     = (float*)alloc((size_t)NSLICE * BB * N0 * 4);    // split-K partials
    float* z1     = (float*)alloc((size_t)NSLICE * BB * N0 * 4);
    float* c0     = (float*)alloc((size_t)BB   * HID * 4);
    float* c1     = (float*)alloc((size_t)BB   * HID * 4);
    bf16*  hcat   = (bf16*) alloc((size_t)BB   * KP2 * 2);          // [h0 | h1] padded
    bf16*  outsb  = (bf16*) alloc((size_t)ROWS * KP  * 2);          // h1 over time
    float* logits = (float*)alloc((size_t)ROWS * NV  * 4);
    float* rloss  = (float*)alloc((size_t)ROWS * 4);

    size_t used = off;
    if (used > ws_size) used = ws_size;
    hipMemsetAsync(d_ws, 0, used, stream);   // zeros padding + initial c/h state

    // weight conversion to bf16, padded
    convert_pad<<<1500, 256, 0, stream>>>(W0,  6000,    0,  6000, W0xb, N0,    0);
    convert_pad<<<1500, 256, 0, stream>>>(W0,  6000, 1500,  6000, W0hb, N0,    0);
    convert_pad<<<1500, 256, 0, stream>>>(W1,  6000,    0,  6000, W1cb, N0,    0);
    convert_pad<<<1500, 256, 0, stream>>>(W1,  6000, 1500,  6000, W1cb, N0, 1504);
    convert_pad<<<1500, 256, 0, stream>>>(Wsm,VOCAB,    0, VOCAB, Wsmb, NV,    0);

    embed_kernel<<<ROWS, 256, 0, stream>>>(x, emb, Xb);

    // parallel: xW0 = x_emb @ W0[:EMB]   (2240 x 1504 x 6144), single K slice
    gemm_bf16_wmma<<<dim3(N0/256, ROWS/64, 1), 256, 0, stream>>>(
        Xb, KP, W0xb, N0, xW0, N0, KP, KP/32, 0);

    const size_t zsl = (size_t)BB * N0;
    // sequential recurrence; recurrent GEMMs use split-K=4 for occupancy
    for (int t = 0; t < TT; ++t) {
        // z0 partials = h0 @ W0h   (A = hcat[:, :KP], lda = KP2), 47 tiles -> 12/slice
        gemm_bf16_wmma<<<dim3(N0/256, 1, NSLICE), 256, 0, stream>>>(
            hcat, KP2, W0hb, N0, z0, N0, KP, 12, zsl);
        lstm_cell<<<(BB*HID + 255)/256, 256, 0, stream>>>(
            z0, N0, NSLICE, zsl, xW0 + (size_t)t * BB * N0, N0, b0, c0,
            hcat, KP2, (bf16*)nullptr, 0);
        // z1 partials = [h0;h1] @ W1   (K=3008), 94 tiles -> 24/slice
        gemm_bf16_wmma<<<dim3(N0/256, 1, NSLICE), 256, 0, stream>>>(
            hcat, KP2, W1cb, N0, z1, N0, KP2, 24, zsl);
        lstm_cell<<<(BB*HID + 255)/256, 256, 0, stream>>>(
            z1, N0, NSLICE, zsl, (const float*)nullptr, 0, b1, c1,
            hcat + KP, KP2, outsb + (size_t)t * BB * KP, KP);
    }

    // logits = outputs @ softmax_w   (2240 x 1504 x 10240), single K slice
    gemm_bf16_wmma<<<dim3(NV/256, ROWS/64, 1), 256, 0, stream>>>(
        outsb, KP, Wsmb, NV, logits, NV, KP, KP/32, 0);

    loss_row_kernel<<<ROWS, 256, 0, stream>>>(logits, NV, bsm, y, rloss);
    final_reduce<<<1, 256, 0, stream>>>(rloss, out);

    (void)in_sizes; (void)n_in; (void)out_size;
}
